// QuantumLayer_13709535609621
// MI455X (gfx1250) — matrix operand order repaired and checked by reading, not compiled, not run on hardware
//
#include <hip/hip_runtime.h>
#include <hip/hip_bf16.h>

typedef _Float16 v16h __attribute__((ext_vector_type(16)));
typedef _Float16 v8h  __attribute__((ext_vector_type(8)));
typedef _Float16 v4h  __attribute__((ext_vector_type(4)));
typedef float    v8f  __attribute__((ext_vector_type(8)));
typedef unsigned int u32x4 __attribute__((ext_vector_type(4)));
typedef int          i32x8 __attribute__((ext_vector_type(8)));
typedef int          i32x4 __attribute__((ext_vector_type(4)));

#define N_TOK 32768   // B*S = 16*2048
#define FDIM  128
#define NG    32      // feature groups (n_splits)
#define WROW  136     // LDS row stride in halves after TDM padding (64+4 dwords)

// ---------------------------------------------------------------------------
// Kernel 1: build the fixed 16x16 unitary for each of the 32 groups.
// One thread per (group, basis column): evolve |col> through the data-
// independent part of the circuit (2 layers of RX,RY,RZ per qubit + CNOT
// chain + CZ(3,0)). Store U in layout [row][col][group] (float2) so the
// simulation kernel gets conflict-free LDS reads.
// ---------------------------------------------------------------------------
__global__ void build_unitaries(const float* __restrict__ qw, float2* __restrict__ U)
{
    int tid = threadIdx.x;
    if (tid >= NG * 16) return;
    int j   = tid >> 4;   // group
    int col = tid & 15;   // basis column

    float re[16], im[16];
#pragma unroll
    for (int l = 0; l < 16; ++l) { re[l] = 0.f; im[l] = 0.f; }
    re[col] = 1.f;

    const float* w = qw + j * 8;  // (N_LAYERS=2, N_QUBITS=4)

#pragma unroll
    for (int layer = 0; layer < 2; ++layer) {
#pragma unroll
        for (int q = 0; q < 4; ++q) {
            float th = w[layer * 4 + q];
            float c = cosf(0.5f * th);
            float s = sinf(0.5f * th);
            const int mask = 8 >> q;  // qubit q <-> bit (3-q)
            // RX: [[c, -i s], [-i s, c]]
#pragma unroll
            for (int l = 0; l < 16; ++l) if (!(l & mask)) {
                int l1 = l | mask;
                float r0 = re[l], i0 = im[l], r1 = re[l1], i1 = im[l1];
                re[l]  = c * r0 + s * i1;   im[l]  = c * i0 - s * r1;
                re[l1] = s * i0 + c * r1;   im[l1] = c * i1 - s * r0;
            }
            // RY: [[c, -s], [s, c]] (real)
#pragma unroll
            for (int l = 0; l < 16; ++l) if (!(l & mask)) {
                int l1 = l | mask;
                float r0 = re[l], i0 = im[l], r1 = re[l1], i1 = im[l1];
                re[l]  = c * r0 - s * r1;   im[l]  = c * i0 - s * i1;
                re[l1] = s * r0 + c * r1;   im[l1] = s * i0 + c * i1;
            }
            // RZ: diag(c - i s, c + i s)
#pragma unroll
            for (int l = 0; l < 16; ++l) {
                float r = re[l], i2 = im[l];
                if (l & mask) { re[l] = c * r - s * i2; im[l] = c * i2 + s * r; }
                else          { re[l] = c * r + s * i2; im[l] = c * i2 - s * r; }
            }
        }
        // CNOT chain: control q, target q+1  (swap within control=1 subspace)
#pragma unroll
        for (int q = 0; q < 3; ++q) {
            const int cm = 8 >> q, tm = 8 >> (q + 1);
#pragma unroll
            for (int l = 0; l < 16; ++l)
                if ((l & cm) && !(l & tm)) {
                    int l1 = l | tm;
                    float tr = re[l], ti = im[l];
                    re[l] = re[l1]; im[l] = im[l1];
                    re[l1] = tr;    im[l1] = ti;
                }
        }
        // CZ(qubit3, qubit0): -1 where both bits set
#pragma unroll
        for (int l = 0; l < 16; ++l)
            if ((l & 8) && (l & 1)) { re[l] = -re[l]; im[l] = -im[l]; }
    }

#pragma unroll
    for (int r = 0; r < 16; ++r)
        U[(r * 16 + col) * NG + j] = make_float2(re[r], im[r]);
}

// ---------------------------------------------------------------------------
// Kernel 2: convert W1, W2 to f16 for the WMMA GEMMs.
// ---------------------------------------------------------------------------
__global__ void convert_w(const float* __restrict__ W1, const float* __restrict__ W2,
                          _Float16* __restrict__ W1h, _Float16* __restrict__ W2h)
{
    int i = blockIdx.x * blockDim.x + threadIdx.x;
    if (i < FDIM * FDIM) {
        W1h[i] = (_Float16)W1[i];
        W2h[i] = (_Float16)W2[i];
    }
}

// ---------------------------------------------------------------------------
// Kernel 3: quantum simulation.  psi = U[j] * v(x), q_i = sum_r sign_i(r)|psi_r|^2
// v is a rank-1 product state: v_l = p_l * (-i)^popcount(l), p_l real products
// of cos/sin half-angles.  Block = 256 threads = 8 tokens x 32 groups.
// All 32 unitaries (64 KB) staged in LDS.
// ---------------------------------------------------------------------------
__global__ void quantum_sim(const float* __restrict__ x,
                            const float2* __restrict__ Ug,
                            _Float16* __restrict__ qh)
{
    __shared__ float2 uS[16 * 16 * NG];   // 64 KB
    const int tid = threadIdx.x;
    {
        const float* up = (const float*)Ug;
        float* us = (float*)uS;
        for (int i = tid; i < 16 * 16 * NG * 2; i += 256) us[i] = up[i];
    }
    __syncthreads();

    const int j     = tid >> 3;                 // group 0..31 (4 groups per wave)
    const int token = blockIdx.x * 8 + (tid & 7);

    const float* xp = x + (size_t)token * FDIM + j * 4;
    float c[4], s[4];
#pragma unroll
    for (int a = 0; a < 4; ++a) {
        float h = 0.5f * xp[a];
        c[a] = cosf(h);
        s[a] = sinf(h);
    }
    float p[16];
#pragma unroll
    for (int l = 0; l < 16; ++l)
        p[l] = ((l & 8) ? s[0] : c[0]) * ((l & 4) ? s[1] : c[1]) *
               ((l & 2) ? s[2] : c[2]) * ((l & 1) ? s[3] : c[3]);

    float z0 = 0.f, z1 = 0.f, z2 = 0.f, z3 = 0.f;
#pragma unroll
    for (int r = 0; r < 16; ++r) {
        float ar = 0.f, ai = 0.f;
#pragma unroll
        for (int l = 0; l < 16; ++l) {
            float2 u = uS[(r * 16 + l) * NG + j];
            float pl = p[l];
            const int ph = __builtin_popcount(l) & 3;   // v_l = p_l * (-i)^ph
            if (ph == 0)      { ar += u.x * pl; ai += u.y * pl; }
            else if (ph == 1) { ar += u.y * pl; ai -= u.x * pl; }
            else if (ph == 2) { ar -= u.x * pl; ai -= u.y * pl; }
            else              { ar -= u.y * pl; ai += u.x * pl; }
        }
        float prob = ar * ar + ai * ai;
        z0 += (r & 8) ? -prob : prob;
        z1 += (r & 4) ? -prob : prob;
        z2 += (r & 2) ? -prob : prob;
        z3 += (r & 1) ? -prob : prob;
    }

    v4h o;
    o[0] = (_Float16)z0; o[1] = (_Float16)z1;
    o[2] = (_Float16)z2; o[3] = (_Float16)z3;
    *(v4h*)(qh + (size_t)token * FDIM + j * 4) = o;   // 8-byte coalesced store
}

// ---------------------------------------------------------------------------
// TDM: DMA the full 128x128 f16 weight matrix from global into LDS.
// D# per CDNA5 ISA:
//   group0: count=1, lds_addr, global_addr[56:0], type=2
//   group1: data_size=2B, tensor 128x128, tile 128x128, dim0_stride=128,
//           pad_enable, pad_interval=5 (every 64 dwords = one row),
//           pad_amount=3 (+4 dwords) -> LDS row stride 68 dwords (136 halves)
//           so B-frag ds_load_b128 across 16 lanes is bank-conflict-free.
// Groups 2/3 unused (2D tile: tile_dim2..4 = 0); extra i32x8 slot of the
// 6-arg clang-23 builtin is zero-filled.
// Caller: one wave issues + waits TENSORcnt, then workgroup barrier.
// ---------------------------------------------------------------------------
__device__ __forceinline__ void tdm_load_w(const _Float16* W, _Float16* lds)
{
    unsigned long long ga = (unsigned long long)(size_t)W;
    unsigned int laddr = (unsigned int)(size_t)lds;   // flat addr low 32 = LDS offset

    u32x4 g0;
    g0[0] = 1u;                                        // count=1 (valid user D#)
    g0[1] = laddr;                                     // lds_addr
    g0[2] = (unsigned int)(ga & 0xFFFFFFFFu);          // global_addr[31:0]
    g0[3] = (unsigned int)((ga >> 32) & 0x01FFFFFFu)   // global_addr[56:32]
          | (2u << 30);                                // type=2 ("image")

    i32x8 g1;
    g1[0] = (int)((1u << 16)      // data_size = 1 -> 2 bytes
                | (1u << 20)      // pad_enable
                | (5u << 22)      // pad_interval: 2^(5+1)=64 dwords
                | (3u << 25));    // pad_amount: 3+1 = 4 dwords
    g1[1] = (int)(128u << 16);    // tensor_dim0[15:0] at bits[63:48]
    g1[2] = (int)(128u << 16);    // tensor_dim0 hi=0 | tensor_dim1[15:0]
    g1[3] = (int)(128u << 16);    // tensor_dim1 hi=0 | tile_dim0=128
    g1[4] = 128;                  // tile_dim1=128, tile_dim2=0
    g1[5] = 128;                  // tensor_dim0_stride low32 = 128
    g1[6] = 0;
    g1[7] = 0;

    i32x4 gz4 = {0, 0, 0, 0};             // groups 2/3 unused (2D tile)
    i32x8 gz8 = {0, 0, 0, 0, 0, 0, 0, 0}; // uncharacterized extra slot
    __builtin_amdgcn_tensor_load_to_lds(g0, g1, gz4, gz4, gz8, 0);
    __builtin_amdgcn_s_wait_tensorcnt(0);
}

// ---------------------------------------------------------------------------
// Kernels 4/5: WMMA GEMM  out[t][n] = act( sum_k A[t][k] * W[n][k] + bias[n] )
// 256 threads = 8 waves; each wave owns one 16x16 output tile (8 col tiles),
// block owns 16 rows; K=128 -> 4x V_WMMA_F32_16X16X32_F16.
// A-frag (global): lane m loads K {kb..kb+7} and {kb+16..kb+23}, kb=8*(lane>=16)
// B-frag (LDS via TDM): lane col=lane&15 loads K {kb..kb+7} and {kb+16..kb+23} from W row n.
// C/D:    VGPR g -> row g (lanes 0-15) / g+8 (lanes 16-31), col = lane&15.
// ---------------------------------------------------------------------------
template <int RELU>
__global__ void mlp_gemm(const _Float16* __restrict__ A,
                         const _Float16* __restrict__ W,
                         const float* __restrict__ bias,
                         _Float16* __restrict__ outH,
                         float* __restrict__ outF)
{
    __shared__ __align__(16) _Float16 Wl[FDIM * WROW + 8];  // ~34.8 KB padded
    if (threadIdx.x < 32) tdm_load_w(W, Wl);   // wave 0 DMAs + waits TENSORcnt
    __syncthreads();

    const int lane = threadIdx.x & 31;
    const int wv   = threadIdx.x >> 5;       // 0..7 -> column tile
    const int row0 = blockIdx.x * 16;
    const int n0   = wv * 16;
    const int m    = lane & 15;
    const int hi   = lane >> 4;

    v8f acc = {};
#pragma unroll
    for (int kt = 0; kt < 4; ++kt) {
        const int kb  = kt * 32 + hi * 8;    // A chunks
        const int kb2 = kt * 32 + hi * 16;   // unused; B also loads at kb
        union { v16h v; v8h h[2]; } a, b;
        a.h[0] = *(const v8h*)(A + (size_t)(row0 + m) * FDIM + kb);
        a.h[1] = *(const v8h*)(A + (size_t)(row0 + m) * FDIM + kb + 16);
        b.h[0] = *(const v8h*)(Wl + (n0 + m) * WROW + kb);
        b.h[1] = *(const v8h*)(Wl + (n0 + m) * WROW + kb + 16);
        acc = __builtin_amdgcn_wmma_f32_16x16x32_f16(
                  false, a.v, false, b.v, (short)0, acc, false, false);
    }

    const int col = n0 + m;
    const float bv = bias[col];
#pragma unroll
    for (int g = 0; g < 8; ++g) {
        const int row = row0 + g + hi * 8;
        float v = acc[g] + bv;
        if (RELU) {
            v = v > 0.f ? v : 0.f;
            outH[(size_t)row * FDIM + col] = (_Float16)v;
        } else {
            outF[(size_t)row * FDIM + col] = v;
        }
    }
}

// ---------------------------------------------------------------------------
extern "C" void kernel_launch(void* const* d_in, const int* in_sizes, int n_in,
                              void* d_out, int out_size, void* d_ws, size_t ws_size,
                              hipStream_t stream)
{
    const float* x  = (const float*)d_in[0];   // (16,2048,128)
    const float* qw = (const float*)d_in[1];   // (32,2,4)
    const float* W1 = (const float*)d_in[2];   // (128,128)
    const float* b1 = (const float*)d_in[3];   // (128)
    const float* W2 = (const float*)d_in[4];   // (128,128)
    const float* b2 = (const float*)d_in[5];   // (128)
    float* out = (float*)d_out;                // (16,2048,128) f32

    char* ws = (char*)d_ws;
    float2*   U   = (float2*)ws;          ws += 16 * 16 * NG * sizeof(float2); // 64 KB
    _Float16* W1h = (_Float16*)ws;        ws += FDIM * FDIM * sizeof(_Float16);
    _Float16* W2h = (_Float16*)ws;        ws += FDIM * FDIM * sizeof(_Float16);
    _Float16* qh  = (_Float16*)ws;        ws += (size_t)N_TOK * FDIM * sizeof(_Float16);
    _Float16* hh  = (_Float16*)ws;        // (size_t)N_TOK * FDIM * sizeof(_Float16)

    build_unitaries<<<1, 512, 0, stream>>>(qw, U);
    convert_w<<<(FDIM * FDIM + 255) / 256, 256, 0, stream>>>(W1, W2, W1h, W2h);
    quantum_sim<<<N_TOK / 8, 256, 0, stream>>>(x, U, qh);
    mlp_gemm<1><<<N_TOK / 16, 256, 0, stream>>>(qh, W1h, b1, hh, nullptr);
    mlp_gemm<0><<<N_TOK / 16, 256, 0, stream>>>(hh, W2h, b2, nullptr, out);
}
